// ImprovedGraphEncoder_82403242541245
// MI455X (gfx1250) — matrix operand order repaired and checked by reading, not compiled, not run on hardware
//
#include <hip/hip_runtime.h>
#include <hip/hip_bf16.h>
#include <math.h>

#define NN 100000
#define EE 1600000
#define DD 128

typedef __attribute__((ext_vector_type(16))) _Float16 v16h;
typedef __attribute__((ext_vector_type(8)))  float    v8f;

struct alignas(16) U128 { unsigned int u[4]; };

__device__ __forceinline__ float gelu_exact(float x) {
    return 0.5f * x * (1.0f + erff(x * 0.70710678118654752440f));
}

__device__ __forceinline__ void atomicMaxF(float* addr, float v) {
    if (v >= 0.0f) atomicMax((int*)addr, __float_as_int(v));
    else           atomicMin((unsigned int*)addr, __float_as_uint(v));
}

// ---------------- generic helpers ----------------
__global__ __launch_bounds__(256)
void fill_f32(float* __restrict__ p, float v, size_t n) {
    size_t i = (size_t)blockIdx.x * blockDim.x + threadIdx.x;
    if (i < n) p[i] = v;
}

__global__ __launch_bounds__(256)
void cast_f32_to_f16(const float* __restrict__ in, _Float16* __restrict__ out, size_t n) {
    size_t i = (size_t)blockIdx.x * blockDim.x + threadIdx.x;
    if (i < n) out[i] = (_Float16)in[i];
}

// ---------------- WMMA GEMM: C[nrows,128] = A[nrows,128] @ W[128,128]^T (+bias) (+= if accumulate) ----
// block = 128 threads (4 waves); each wave computes a 16x16 tile; block covers 16 rows x 64 cols.
__global__ __launch_bounds__(128)
void gemm128_wmma(const _Float16* __restrict__ A,
                  const _Float16* __restrict__ W,
                  const float* __restrict__ bias,
                  float* __restrict__ C,
                  int nrows, int accumulate)
{
    const int wave = threadIdx.x >> 5;
    const int lane = threadIdx.x & 31;
    const int row0 = blockIdx.x << 4;
    const int col0 = (blockIdx.y << 6) + (wave << 4);
    const int r  = lane & 15;
    const int hh = lane >> 4;

    v8f acc = {};
    const _Float16* ap = A + (size_t)(row0 + r) * DD + hh * 8;   // A frag rows
    const _Float16* bp = W + (size_t)(col0 + r) * DD + hh * 8;   // B frag = W rows (col-major B)
#pragma unroll
    for (int k0 = 0; k0 < DD; k0 += 32) {
        union { U128 q[2]; v16h v; } a;
        union { U128 q[2]; v16h v; } b;
        a.q[0] = *(const U128*)(ap + k0);        // K = k0 + hh*8 .. +7
        a.q[1] = *(const U128*)(ap + k0 + 16);   // K = k0+16 + hh*8 .. +7
        b.q[0] = *(const U128*)(bp + k0);        // K = k0 + hh*8 .. +7
        b.q[1] = *(const U128*)(bp + k0 + 16);   // K = k0+16 + hh*8 .. +7
        acc = __builtin_amdgcn_wmma_f32_16x16x32_f16(false, a.v, false, b.v,
                                                     (short)0, acc, false, false);
    }
    const int cj = col0 + r;
    const float bv = bias ? bias[cj] : 0.0f;
    float* cp = C + (size_t)(row0 + (hh << 3)) * DD + cj;
#pragma unroll
    for (int v = 0; v < 8; ++v) {
        float y = acc[v] + bv;
        if (accumulate) y += cp[(size_t)v * DD];
        cp[(size_t)v * DD] = y;
    }
}

// ---------------- fused (pre-bias) + LayerNorm + GELU + residual + f16 recast ----------------
// one wave per row of 128; 8 rows per 256-thread block
__global__ __launch_bounds__(256)
void ln_gelu_f32(const float* __restrict__ in,
                 const float* __restrict__ pre_bias,
                 const float* __restrict__ gamma,
                 const float* __restrict__ beta,
                 const float* __restrict__ residual,
                 float* __restrict__ out_f32,
                 _Float16* __restrict__ out_f16,
                 int nrows)
{
    const int wave = threadIdx.x >> 5;
    const int lane = threadIdx.x & 31;
    const int row  = blockIdx.x * 8 + wave;
    if (row >= nrows) return;
    const size_t base = (size_t)row * DD;
    float v[4];
    float s = 0.0f;
#pragma unroll
    for (int i = 0; i < 4; ++i) {
        const int c = lane + (i << 5);
        float x = in[base + c];
        if (pre_bias) x += pre_bias[c];
        v[i] = x; s += x;
    }
#pragma unroll
    for (int m = 16; m > 0; m >>= 1) s += __shfl_xor(s, m, 32);
    const float mean = s * (1.0f / DD);
    float ss = 0.0f;
#pragma unroll
    for (int i = 0; i < 4; ++i) { const float d = v[i] - mean; ss += d * d; }
#pragma unroll
    for (int m = 16; m > 0; m >>= 1) ss += __shfl_xor(ss, m, 32);
    const float rstd = rsqrtf(ss * (1.0f / DD) + 1e-5f);
#pragma unroll
    for (int i = 0; i < 4; ++i) {
        const int c = lane + (i << 5);
        float y = (v[i] - mean) * rstd * gamma[c] + beta[c];
        y = gelu_exact(y);
        if (residual) y += residual[base + c];
        if (out_f32)  out_f32[base + c] = y;
        if (out_f16)  out_f16[base + c] = (_Float16)y;
    }
}

// ---------------- SAGE mean-aggregation scatter: wave per edge, float4 per lane ----------------
__global__ __launch_bounds__(256)
void sage_scatter(const int* __restrict__ src, const int* __restrict__ dst,
                  const float* __restrict__ h, float* __restrict__ agg,
                  float* __restrict__ deg, int nedges)
{
    const int wave = threadIdx.x >> 5;
    const int lane = threadIdx.x & 31;
    const int e = blockIdx.x * 8 + wave;
    if (e >= nedges) return;
    const int s = src[e], d = dst[e];
    const float4 vs = *(const float4*)(h + (size_t)s * DD + (lane << 2));
    float* dp = agg + (size_t)d * DD + (lane << 2);
    atomicAdd(dp + 0, vs.x);
    atomicAdd(dp + 1, vs.y);
    atomicAdd(dp + 2, vs.z);
    atomicAdd(dp + 3, vs.w);
    if (lane == 0) atomicAdd(deg + d, 1.0f);
}

// agg = agg / max(deg,1), cast to f16
__global__ __launch_bounds__(256)
void agg_norm_cast(const float* __restrict__ agg, const float* __restrict__ deg,
                   _Float16* __restrict__ out, size_t n)
{
    size_t i = (size_t)blockIdx.x * blockDim.x + threadIdx.x;
    if (i >= n) return;
    const int node = (int)(i >> 7);
    out[i] = (_Float16)(agg[i] / fmaxf(deg[node], 1.0f));
}

// ---------------- GATv2 ----------------
// edge score: e = leaky_relu(xl[s]+xr[d], 0.2) . att   (wave per edge incl. self loops)
__global__ __launch_bounds__(256)
void gat_edge_score(const int* __restrict__ src, const int* __restrict__ dst,
                    const float* __restrict__ xl, const float* __restrict__ xr,
                    const float* __restrict__ att, float* __restrict__ e_out,
                    int nedges, int ntot)
{
    const int wave = threadIdx.x >> 5;
    const int lane = threadIdx.x & 31;
    const int i = blockIdx.x * 8 + wave;
    if (i >= ntot) return;
    int s, d;
    if (i < nedges) { s = src[i]; d = dst[i]; } else { s = d = i - nedges; }
    float acc = 0.0f;
#pragma unroll
    for (int k = 0; k < 4; ++k) {
        const int c = lane + (k << 5);
        float v = xl[(size_t)s * DD + c] + xr[(size_t)d * DD + c];
        v = (v > 0.0f) ? v : 0.2f * v;
        acc += v * att[c];
    }
#pragma unroll
    for (int m = 16; m > 0; m >>= 1) acc += __shfl_xor(acc, m, 32);
    if (lane == 0) e_out[i] = acc;
}

__global__ __launch_bounds__(256)
void gat_seg_max(const int* __restrict__ dst, const float* __restrict__ e,
                 float* __restrict__ m, int nedges, int ntot)
{
    const int i = blockIdx.x * blockDim.x + threadIdx.x;
    if (i >= ntot) return;
    const int d = (i < nedges) ? dst[i] : (i - nedges);
    atomicMaxF(&m[d], e[i]);
}

__global__ __launch_bounds__(256)
void gat_exp_sum(const int* __restrict__ dst, float* __restrict__ e,
                 const float* __restrict__ m, float* __restrict__ z,
                 int nedges, int ntot)
{
    const int i = blockIdx.x * blockDim.x + threadIdx.x;
    if (i >= ntot) return;
    const int d = (i < nedges) ? dst[i] : (i - nedges);
    const float v = expf(e[i] - m[d]);
    e[i] = v;
    atomicAdd(&z[d], v);
}

__global__ __launch_bounds__(256)
void gat_scatter(const int* __restrict__ src, const int* __restrict__ dst,
                 const float* __restrict__ e, const float* __restrict__ z,
                 const float* __restrict__ xl, float* __restrict__ gout,
                 int nedges, int ntot)
{
    const int wave = threadIdx.x >> 5;
    const int lane = threadIdx.x & 31;
    const int i = blockIdx.x * 8 + wave;
    if (i >= ntot) return;
    int s, d;
    if (i < nedges) { s = src[i]; d = dst[i]; } else { s = d = i - nedges; }
    const float alpha = e[i] / z[d];
    const float4 vx = *(const float4*)(xl + (size_t)s * DD + (lane << 2));
    float* gp = gout + (size_t)d * DD + (lane << 2);
    atomicAdd(gp + 0, alpha * vx.x);
    atomicAdd(gp + 1, alpha * vx.y);
    atomicAdd(gp + 2, alpha * vx.z);
    atomicAdd(gp + 3, alpha * vx.w);
}

// ---------------- final row L2 normalize ----------------
__global__ __launch_bounds__(256)
void l2norm_rows(const float* __restrict__ in, float* __restrict__ out, int nrows)
{
    const int wave = threadIdx.x >> 5;
    const int lane = threadIdx.x & 31;
    const int row  = blockIdx.x * 8 + wave;
    if (row >= nrows) return;
    const size_t base = (size_t)row * DD;
    float v[4];
    float s = 0.0f;
#pragma unroll
    for (int i = 0; i < 4; ++i) {
        v[i] = in[base + lane + (i << 5)];
        s += v[i] * v[i];
    }
#pragma unroll
    for (int m = 16; m > 0; m >>= 1) s += __shfl_xor(s, m, 32);
    const float inv = 1.0f / fmaxf(sqrtf(s), 1e-12f);
#pragma unroll
    for (int i = 0; i < 4; ++i) out[base + lane + (i << 5)] = v[i] * inv;
}

// ============================================================
extern "C" void kernel_launch(void* const* d_in, const int* in_sizes, int n_in,
                              void* d_out, int out_size, void* d_ws, size_t ws_size,
                              hipStream_t stream)
{
    const int N = NN, E = EE;
    const int M = E + N;                 // GAT edges incl. self loops
    const size_t ND = (size_t)N * DD;
    const size_t WW = (size_t)DD * DD;

    const float* x        = (const float*)d_in[0];
    const int*   ei       = (const int*)d_in[1];
    const int*   src      = ei;
    const int*   dst      = ei + E;
    const float* w_in     = (const float*)d_in[2];
    const float* b_in     = (const float*)d_in[3];
    const float* g_in     = (const float*)d_in[4];
    const float* beta_in  = (const float*)d_in[5];
    const float* sage_wl  = (const float*)d_in[6];
    const float* sage_bl  = (const float*)d_in[7];
    const float* sage_wr  = (const float*)d_in[8];
    const float* g1       = (const float*)d_in[9];
    const float* beta1    = (const float*)d_in[10];
    const float* gat_wl   = (const float*)d_in[11];
    const float* gat_bl   = (const float*)d_in[12];
    const float* gat_wr   = (const float*)d_in[13];
    const float* gat_br   = (const float*)d_in[14];
    const float* gat_att  = (const float*)d_in[15];
    const float* gat_bias = (const float*)d_in[16];
    const float* g2       = (const float*)d_in[17];
    const float* beta2    = (const float*)d_in[18];
    const float* w_o1     = (const float*)d_in[19];
    const float* b_o1     = (const float*)d_in[20];
    const float* g_o      = (const float*)d_in[21];
    const float* beta_o   = (const float*)d_in[22];
    const float* w_o2     = (const float*)d_in[23];
    const float* b_o2     = (const float*)d_in[24];

    // ---- workspace carve-out ----
    char* p = (char*)d_ws;
    float* t    = (float*)p; p += ND * 4;     // GEMM scratch
    float* u    = (float*)p; p += ND * 4;     // GEMM scratch 2
    float* h0   = (float*)p; p += ND * 4;
    float* h1   = (float*)p; p += ND * 4;
    float* h2   = (float*)p; p += ND * 4;
    float* xl   = (float*)p; p += ND * 4;
    float* xr   = (float*)p; p += ND * 4;
    float* gout = (float*)p; p += ND * 4;
    float* deg  = (float*)p; p += (size_t)N * 4;
    float* mmax = (float*)p; p += (size_t)N * 4;
    float* zsum = (float*)p; p += (size_t)N * 4;
    float* ebuf = (float*)p; p += (size_t)M * 4;
    _Float16* x_h   = (_Float16*)p; p += ND * 2;
    _Float16* h0_h  = (_Float16*)p; p += ND * 2;
    _Float16* h1_h  = (_Float16*)p; p += ND * 2;
    _Float16* h2_h  = (_Float16*)p; p += ND * 2;
    _Float16* agg_h = (_Float16*)p; p += ND * 2;
    _Float16* t3_h  = (_Float16*)p; p += ND * 2;
    _Float16* wh[7];
    for (int i = 0; i < 7; ++i) { wh[i] = (_Float16*)p; p += WW * 2; }

    const dim3 gemm_grid(N / 16, 2);
    const int row_blocks  = (N + 7) / 8;            // 8 rows per 256-thr block
    const int edge_waves  = (E + 7) / 8;
    const int gat_waves   = (M + 7) / 8;
    const int gat_threads = (M + 255) / 256;

    // f16 casts of x and all weights
    cast_f32_to_f16<<<(int)((ND + 255) / 256), 256, 0, stream>>>(x, x_h, ND);
    const float* wsrc[7] = { w_in, sage_wl, sage_wr, gat_wl, gat_wr, w_o1, w_o2 };
    for (int i = 0; i < 7; ++i)
        cast_f32_to_f16<<<(int)((WW + 255) / 256), 256, 0, stream>>>(wsrc[i], wh[i], WW);

    // ---- stage 0: h0 = gelu(ln(x @ w_in.T + b_in)) ----
    gemm128_wmma<<<gemm_grid, 128, 0, stream>>>(x_h, wh[0], b_in, t, N, 0);
    ln_gelu_f32<<<row_blocks, 256, 0, stream>>>(t, nullptr, g_in, beta_in, nullptr, h0, h0_h, N);

    // ---- stage 1: SAGE ----
    fill_f32<<<(int)((ND + 255) / 256), 256, 0, stream>>>(t, 0.0f, ND);
    fill_f32<<<(N + 255) / 256, 256, 0, stream>>>(deg, 0.0f, (size_t)N);
    sage_scatter<<<edge_waves, 256, 0, stream>>>(src, dst, h0, t, deg, E);
    agg_norm_cast<<<(int)((ND + 255) / 256), 256, 0, stream>>>(t, deg, agg_h, ND);
    gemm128_wmma<<<gemm_grid, 128, 0, stream>>>(agg_h, wh[1], sage_bl, u, N, 0);
    gemm128_wmma<<<gemm_grid, 128, 0, stream>>>(h0_h, wh[2], nullptr, u, N, 1);
    ln_gelu_f32<<<row_blocks, 256, 0, stream>>>(u, nullptr, g1, beta1, h0, h1, h1_h, N);

    // ---- stage 2: GATv2 ----
    gemm128_wmma<<<gemm_grid, 128, 0, stream>>>(h1_h, wh[3], gat_bl, xl, N, 0);
    gemm128_wmma<<<gemm_grid, 128, 0, stream>>>(h1_h, wh[4], gat_br, xr, N, 0);
    gat_edge_score<<<gat_waves, 256, 0, stream>>>(src, dst, xl, xr, gat_att, ebuf, E, M);
    fill_f32<<<(N + 255) / 256, 256, 0, stream>>>(mmax, -INFINITY, (size_t)N);
    fill_f32<<<(N + 255) / 256, 256, 0, stream>>>(zsum, 0.0f, (size_t)N);
    fill_f32<<<(int)((ND + 255) / 256), 256, 0, stream>>>(gout, 0.0f, ND);
    gat_seg_max<<<gat_threads, 256, 0, stream>>>(dst, ebuf, mmax, E, M);
    gat_exp_sum<<<gat_threads, 256, 0, stream>>>(dst, ebuf, mmax, zsum, E, M);
    gat_scatter<<<gat_waves, 256, 0, stream>>>(src, dst, ebuf, zsum, xl, gout, E, M);
    ln_gelu_f32<<<row_blocks, 256, 0, stream>>>(gout, gat_bias, g2, beta2, h1, h2, h2_h, N);

    // ---- stage 3: output MLP + row-normalize ----
    gemm128_wmma<<<gemm_grid, 128, 0, stream>>>(h2_h, wh[5], b_o1, t, N, 0);
    ln_gelu_f32<<<row_blocks, 256, 0, stream>>>(t, nullptr, g_o, beta_o, nullptr, nullptr, t3_h, N);
    gemm128_wmma<<<gemm_grid, 128, 0, stream>>>(t3_h, wh[6], b_o2, t, N, 0);
    l2norm_rows<<<row_blocks, 256, 0, stream>>>(t, (float*)d_out, N);
}
